// CovarianceSimilarity_9199819948069
// MI455X (gfx1250) — matrix operand order repaired and checked by reading, not compiled, not run on hardware
//
#include <hip/hip_runtime.h>
#include <hip/hip_bf16.h>

// ---------------------------------------------------------------------------
// CovarianceSimilarity for MI455X (gfx1250, wave32, WMMA bf16 16x16x32)
//
// scores[n] = (1/d) * <cov, q_n q_n^T>   with cov, Gram both computed as
// X @ X^T GEMMs so every WMMA operand is a contiguous row-segment load.
// Symmetry halves both GEMMs (upper-triangular blocks, x2 off-diag weight).
// ---------------------------------------------------------------------------

typedef __attribute__((ext_vector_type(16))) __bf16 v16bf;
typedef __attribute__((ext_vector_type(8)))  __bf16 bf16x8;
typedef __attribute__((ext_vector_type(8)))  float  v8f;

#define C_DIM   1024
#define NSUP    25600      // 25 * 32 * 32
#define D_DIM   1024       // 32 * 32 spatial
#define NQ      64
#define NB32    32         // 1024/32 blocks per dim (cov kernel, 32x32 blocks)
#define NTRI32  528        // NB32*(NB32+1)/2
#define NB64    16         // 1024/64 blocks per dim (score kernel, 64x64 blocks)
#define NTRI64  136        // NB64*(NB64+1)/2
#define EPSF    1e-8f

struct Frag {
    union { v16bf v; bf16x8 h[2]; };
};

__device__ __forceinline__ v8f wmma_bf16(const Frag& a, const Frag& b, v8f c) {
    return __builtin_amdgcn_wmma_f32_16x16x32_bf16(
        /*neg_a=*/false, a.v, /*neg_b=*/false, b.v,
        /*c_mod=*/(short)0, c, /*reuse_a=*/false, /*reuse_b=*/false);
}

// A-fragment (16x32 bf16, M x K), X row-major with leading dim ld:
//   lanes 0-15:  M=lane,    K = kb+{0..7} (v0-3),  kb+{16..23} (v4-7)
//   lanes 16-31: M=lane-16, K = kb+{8..15}(v0-3),  kb+{24..31} (v4-7)
__device__ __forceinline__ void load_a_frag(Frag& f, const __bf16* X, long ld,
                                            int lane, int kb, int row16) {
    int half = lane >> 4;
    int m = row16 + (lane & 15);
    const __bf16* p = X + (long)m * ld + kb + half * 8;
    f.h[0] = *(const bf16x8*)(p);
    f.h[1] = *(const bf16x8*)(p + 16);
}

// B-fragment (32x16 bf16, K x N) of B = X^T  =>  B[k,n] = X[n,k]:
//   lanes 0-15:  N=lane,    K = kb+{0..7} (v0-3),  kb+{16..23} (v4-7)
//   lanes 16-31: N=lane-16, K = kb+{8..15}(v0-3),  kb+{24..31} (v4-7)
__device__ __forceinline__ void load_b_frag(Frag& f, const __bf16* X, long ld,
                                            int lane, int kb, int col16) {
    int half = lane >> 4;
    int n = col16 + (lane & 15);
    const __bf16* p = X + (long)n * ld + kb + half * 8;
    f.h[0] = *(const bf16x8*)(p);
    f.h[1] = *(const bf16x8*)(p + 16);
}

// linear t -> upper-triangular (bm <= bn) block coordinates over NB blocks
__device__ __forceinline__ void tri_map(int t, int NB, int& bm, int& bn) {
    int r = 0;
    while (t >= NB - r) { t -= NB - r; ++r; }
    bm = r; bn = r + t;
}

// ---------------------------------------------------------------------------
// Phase 1: per-channel mean of support over B*h*w = 25600
// ---------------------------------------------------------------------------
__global__ __launch_bounds__(256) void mean_kernel(const float* __restrict__ sup,
                                                   float* __restrict__ mean) {
    int c = blockIdx.x, tid = threadIdx.x;
    float s = 0.f;
    for (int i = tid; i < NSUP; i += 256) {
        int b = i >> 10, sp = i & 1023;
        s += sup[(long)b * (C_DIM * D_DIM) + (long)c * D_DIM + sp];
    }
    __shared__ float red[256];
    red[tid] = s; __syncthreads();
    for (int off = 128; off; off >>= 1) {
        if (tid < off) red[tid] += red[tid + off];
        __syncthreads();
    }
    if (tid == 0) mean[c] = red[0] * (1.0f / (float)NSUP);
}

// ---------------------------------------------------------------------------
// Phase 2: centered support, bf16, layout (C, NSUP) row-major
// ---------------------------------------------------------------------------
__global__ __launch_bounds__(256) void center_kernel(const float* __restrict__ sup,
                                                     const float* __restrict__ mean,
                                                     __bf16* __restrict__ Sc) {
    long idx = (long)blockIdx.x * 256 + threadIdx.x;
    if (idx >= (long)C_DIM * NSUP) return;
    int c  = (int)(idx / NSUP);
    int nb = (int)(idx - (long)c * NSUP);
    int b = nb >> 10, sp = nb & 1023;
    float v = sup[(long)b * (C_DIM * D_DIM) + (long)c * D_DIM + sp] - mean[c];
    Sc[idx] = (__bf16)v;
}

// ---------------------------------------------------------------------------
// Phase 3: cov = Sc @ Sc^T / (N-1+eps), upper-triangular 32x32 blocks,
// one wave per block, 2x2 WMMA tiles, K = 25600 (800 k-steps).
// (2x2 keeps 528 independent waves for occupancy; cov is ~30% of FLOPs.)
// ---------------------------------------------------------------------------
__global__ __launch_bounds__(32) void cov_kernel(const __bf16* __restrict__ Sc,
                                                 float* __restrict__ cov) {
    int bm, bn; tri_map(blockIdx.x, NB32, bm, bn);
    int lane = threadIdx.x;
    const long ld = NSUP;
    v8f acc[2][2] = {};
    int r0 = bm * 32, r1 = bm * 32 + 16;
    int c0 = bn * 32, c1 = bn * 32 + 16;
    for (int kb = 0; kb < NSUP; kb += 32) {
        Frag a0, a1, b0, b1;
        load_a_frag(a0, Sc, ld, lane, kb, r0);
        load_a_frag(a1, Sc, ld, lane, kb, r1);
        load_b_frag(b0, Sc, ld, lane, kb, c0);
        load_b_frag(b1, Sc, ld, lane, kb, c1);
        acc[0][0] = wmma_bf16(a0, b0, acc[0][0]);
        acc[0][1] = wmma_bf16(a0, b1, acc[0][1]);
        acc[1][0] = wmma_bf16(a1, b0, acc[1][0]);
        acc[1][1] = wmma_bf16(a1, b1, acc[1][1]);
    }
    const float scale = 1.0f / ((float)(NSUP - 1) + EPSF);
    int half = lane >> 4;
    for (int ti = 0; ti < 2; ++ti)
        for (int tj = 0; tj < 2; ++tj)
            for (int r = 0; r < 8; ++r) {
                int m = bm * 32 + ti * 16 + (half ? r + 8 : r);
                int n = bn * 32 + tj * 16 + (lane & 15);
                cov[(long)m * C_DIM + n] = acc[ti][tj][r] * scale;
            }
}

// ---------------------------------------------------------------------------
// Phase 4: L2-normalize each (n,c) query row over spatial d, emit bf16
// ---------------------------------------------------------------------------
__global__ __launch_bounds__(256) void qnorm_kernel(const float* __restrict__ q,
                                                    __bf16* __restrict__ Qn) {
    long row = blockIdx.x;               // n*C + c, 65536 rows
    const float* src = q + row * D_DIM;
    __bf16* dst = Qn + row * D_DIM;
    int tid = threadIdx.x;
    float s = 0.f;
    for (int j = tid; j < D_DIM; j += 256) { float v = src[j]; s += v * v; }
    __shared__ float red[256];
    red[tid] = s; __syncthreads();
    for (int off = 128; off; off >>= 1) {
        if (tid < off) red[tid] += red[tid + off];
        __syncthreads();
    }
    float rn = 1.0f / (sqrtf(red[0]) + EPSF);
    for (int j = tid; j < D_DIM; j += 256) dst[j] = (__bf16)(src[j] * rn);
}

// ---------------------------------------------------------------------------
// Phase 5: per (n, tri-block): 64x64 G-block = q_n q_n^T via 4x4 WMMA tiles
// (K=1024, 32 k-steps, 16 WMMAs per 8 fragment loads -> 4x operand reuse),
// contract with fp32 cov tiles (x2 strictly-upper), write one partial.
// ---------------------------------------------------------------------------
__global__ __launch_bounds__(32) void score_kernel(const __bf16* __restrict__ Qn,
                                                   const float* __restrict__ cov,
                                                   float* __restrict__ partial) {
    int t  = blockIdx.x % NTRI64;
    int nq = blockIdx.x / NTRI64;
    int bm, bn; tri_map(t, NB64, bm, bn);
    int lane = threadIdx.x;
    const __bf16* X = Qn + (long)nq * C_DIM * D_DIM;
    const long ld = D_DIM;
    v8f acc[4][4] = {};
    for (int kb = 0; kb < D_DIM; kb += 32) {
        Frag a[4], b[4];
#pragma unroll
        for (int i = 0; i < 4; ++i)
            load_a_frag(a[i], X, ld, lane, kb, bm * 64 + i * 16);
#pragma unroll
        for (int j = 0; j < 4; ++j)
            load_b_frag(b[j], X, ld, lane, kb, bn * 64 + j * 16);
#pragma unroll
        for (int i = 0; i < 4; ++i)
#pragma unroll
            for (int j = 0; j < 4; ++j)
                acc[i][j] = wmma_bf16(a[i], b[j], acc[i][j]);
    }
    int half = lane >> 4;
    float s = 0.f;
    for (int ti = 0; ti < 4; ++ti)
        for (int tj = 0; tj < 4; ++tj) {
            int tm = bm * 4 + ti, tn = bn * 4 + tj;
            float f = (tm < tn) ? 2.0f : (tm == tn ? 1.0f : 0.0f);
            if (f == 0.0f) continue;   // lower tiles inside diagonal block
            for (int r = 0; r < 8; ++r) {
                int m = tm * 16 + (half ? r + 8 : r);
                int n = tn * 16 + (lane & 15);
                s += f * acc[ti][tj][r] * cov[(long)m * C_DIM + n];
            }
        }
    __shared__ float red[32];
    red[lane] = s; __syncthreads();
    for (int off = 16; off; off >>= 1) {
        if (lane < off) red[lane] += red[lane + off];
        __syncthreads();
    }
    if (lane == 0) partial[(long)nq * NTRI64 + t] = red[0] * (1.0f / (float)D_DIM);
}

// ---------------------------------------------------------------------------
// Phase 6: deterministic reduction of per-block partials -> scores
// ---------------------------------------------------------------------------
__global__ __launch_bounds__(256) void reduce_kernel(const float* __restrict__ partial,
                                                     float* __restrict__ out) {
    int n = blockIdx.x, tid = threadIdx.x;
    float s = 0.f;
    for (int i = tid; i < NTRI64; i += 256) s += partial[(long)n * NTRI64 + i];
    __shared__ float red[256];
    red[tid] = s; __syncthreads();
    for (int off = 128; off; off >>= 1) {
        if (tid < off) red[tid] += red[tid + off];
        __syncthreads();
    }
    if (tid == 0) out[n] = red[0];
}

// ---------------------------------------------------------------------------
extern "C" void kernel_launch(void* const* d_in, const int* in_sizes, int n_in,
                              void* d_out, int out_size, void* d_ws, size_t ws_size,
                              hipStream_t stream) {
    const float* q   = (const float*)d_in[0];  // (64, 1024, 32, 32)
    const float* sup = (const float*)d_in[1];  // (25, 1024, 32, 32)
    float* out = (float*)d_out;                // (64,)

    char* ws = (char*)d_ws;
    size_t off = 0;
    float*  mean = (float*)(ws + off);  off += 4096;                       // 1024 f32
    __bf16* Sc   = (__bf16*)(ws + off); off += (size_t)C_DIM * NSUP * 2;   // 50 MB
    float*  cov  = (float*)(ws + off);  off += (size_t)C_DIM * C_DIM * 4;  // 4 MB
    __bf16* Qn   = (__bf16*)(ws + off); off += (size_t)NQ * C_DIM * D_DIM * 2; // 128 MB
    float*  part = (float*)(ws + off);  off += (size_t)NQ * NTRI64 * 4;
    (void)in_sizes; (void)n_in; (void)out_size; (void)ws_size;

    mean_kernel<<<C_DIM, 256, 0, stream>>>(sup, mean);

    long tot = (long)C_DIM * NSUP;
    center_kernel<<<(int)((tot + 255) / 256), 256, 0, stream>>>(sup, mean, Sc);

    cov_kernel<<<NTRI32, 32, 0, stream>>>(Sc, cov);

    qnorm_kernel<<<NQ * C_DIM, 256, 0, stream>>>(q, Qn);

    score_kernel<<<NQ * NTRI64, 32, 0, stream>>>(Qn, cov, part);

    reduce_kernel<<<NQ, 256, 0, stream>>>(part, out);
}
